// KoLeoLoss_73538430042938
// MI455X (gfx1250) — matrix operand order repaired and checked by reading, not compiled, not run on hardware
//
#include <hip/hip_runtime.h>
#include <hip/hip_bf16.h>

typedef __attribute__((ext_vector_type(16))) _Float16 v16h;
typedef __attribute__((ext_vector_type(8)))  _Float16 v8h;
typedef __attribute__((ext_vector_type(8)))  float    v8f;
typedef __attribute__((ext_vector_type(4)))  float    v4f;

#define KB_B   32
#define KB_N   1024
#define KB_D   512
#define KB_EPS 1e-8f

__device__ __forceinline__ float wave_sum32(float v) {
#pragma unroll
  for (int off = 16; off >= 1; off >>= 1) v += __shfl_xor(v, off, 32);
  return v;
}

// ---------------------------------------------------------------------------
// Kernel 1: L2-normalize each 512-d vector, emit f16 copy for the WMMA search.
// One wave (32 lanes) per vector, 16 floats per lane.
// ---------------------------------------------------------------------------
__global__ __launch_bounds__(256) void koleo_normalize(const float* __restrict__ x,
                                                       _Float16* __restrict__ xh) {
  const int wave = threadIdx.x >> 5;
  const int lane = threadIdx.x & 31;
  const long vec = (long)blockIdx.x * 8 + wave;

  const float* p = x + vec * KB_D + lane * 16;
  v4f f[4];
#pragma unroll
  for (int t = 0; t < 4; ++t) f[t] = *(const v4f*)(p + 4 * t);

  float ss = 0.0f;
#pragma unroll
  for (int t = 0; t < 4; ++t)
#pragma unroll
    for (int c = 0; c < 4; ++c) ss += f[t][c] * f[t][c];
  ss = wave_sum32(ss);
  const float inv = 1.0f / fmaxf(sqrtf(ss), KB_EPS);

  union { v8h v[2]; _Float16 h[16]; } o;
#pragma unroll
  for (int t = 0; t < 4; ++t)
#pragma unroll
    for (int c = 0; c < 4; ++c) o.h[4 * t + c] = (_Float16)(f[t][c] * inv);

  _Float16* q = xh + vec * KB_D + lane * 16;
  *(v8h*)(q)     = o.v[0];
  *(v8h*)(q + 8) = o.v[1];
}

// ---------------------------------------------------------------------------
// Kernel 2: per-batch similarity GEMM (X * X^T) via v_wmma_f32_16x16x32_f16,
// fused row-wise masked argmax. One workgroup (8 waves) per 16-row tile.
// Each wave sweeps 8 of the 64 column tiles; 16 WMMAs (K=512) per tile.
// ---------------------------------------------------------------------------
__global__ __launch_bounds__(256) void koleo_nn(const _Float16* __restrict__ xh,
                                                int* __restrict__ nnIdx) {
  __shared__ float sval[8][16];
  __shared__ int   sidx[8][16];

  const int wave = threadIdx.x >> 5;
  const int lane = threadIdx.x & 31;
  const int b    = blockIdx.x >> 6;   // batch
  const int rt   = blockIdx.x & 63;   // row tile
  const int R0   = rt * 16;

  const _Float16* Xb = xh + (long)b * KB_N * KB_D;
  const int m  = lane & 15;
  const int hi = lane >> 4;

  // A fragment base: contiguous K (matches B) — lanes 0-15: K = 0..15,
  // lanes 16-31: K = 16..31 (relative to each K0).
  const _Float16* arow = Xb + (long)(R0 + m) * KB_D + hi * 16;

  float best[8];
  int   bidx[8];
#pragma unroll
  for (int r = 0; r < 8; ++r) { best[r] = -3.0f; bidx[r] = 0; }

  for (int ct = wave; ct < 64; ct += 8) {
    const int C0 = ct * 16;
    // B fragment: lane n = m holds 16 contiguous K values of column (C0+n)
    // of X^T, i.e. a contiguous row chunk of X. kbase = hi*16.
    const _Float16* brow = Xb + (long)(C0 + m) * KB_D + hi * 16;

    v8f c = {};
#pragma unroll
    for (int ks = 0; ks < 16; ++ks) {
      const int K0 = ks * 32;
      union { v16h v; v8h h[2]; } a;
      a.h[0] = *(const v8h*)(arow + K0);        // K0 + base + [0..7]
      a.h[1] = *(const v8h*)(arow + K0 + 8);    // K0 + base + [8..15]
      const v16h bm = *(const v16h*)(brow + K0);
      c = __builtin_amdgcn_wmma_f32_16x16x32_f16(
              /*neg_a=*/false, a.v, /*neg_b=*/false, bm,
              /*c_mod=*/(short)0, c, /*reuse_a=*/false, /*reuse_b=*/false);
    }

    const int col = C0 + m;
#pragma unroll
    for (int r = 0; r < 8; ++r) {
      float s = c[r];
      const int row = R0 + hi * 8 + r;
      if (row == col) s = -2.0f;                // mask self-similarity
      if (s > best[r]) { best[r] = s; bidx[r] = col; }
    }
  }

  // Reduce (max, argmax) across the 16 lanes of each half-wave (same row).
#pragma unroll
  for (int r = 0; r < 8; ++r) {
    float v = best[r];
    int   idx = bidx[r];
#pragma unroll
    for (int off = 8; off >= 1; off >>= 1) {
      const float v2 = __shfl_xor(v, off, 32);
      const int   i2 = __shfl_xor(idx, off, 32);
      if (v2 > v || (v2 == v && i2 < idx)) { v = v2; idx = i2; }
    }
    best[r] = v; bidx[r] = idx;
  }

  if (m == 0) {  // lanes 0 and 16 hold rows [R0 + hi*8 + r]
#pragma unroll
    for (int r = 0; r < 8; ++r) {
      sval[wave][hi * 8 + r] = best[r];
      sidx[wave][hi * 8 + r] = bidx[r];
    }
  }
  __syncthreads();

  if (wave == 0 && lane < 16) {
    float v = sval[0][lane];
    int   idx = sidx[0][lane];
#pragma unroll
    for (int w = 1; w < 8; ++w) {
      const float v2 = sval[w][lane];
      const int   i2 = sidx[w][lane];
      if (v2 > v || (v2 == v && i2 < idx)) { v = v2; idx = i2; }
    }
    nnIdx[b * KB_N + R0 + lane] = idx;
  }
}

// ---------------------------------------------------------------------------
// Kernel 3: exact fp32 distance to the nearest neighbor + loss accumulation.
// One wave per row; normalization recomputed inline from the raw input.
// ---------------------------------------------------------------------------
__global__ __launch_bounds__(256) void koleo_loss(const float* __restrict__ x,
                                                  const int* __restrict__ nnIdx,
                                                  float* __restrict__ out) {
  const int wave = threadIdx.x >> 5;
  const int lane = threadIdx.x & 31;
  const long row = (long)blockIdx.x * 8 + wave;
  const long b   = row >> 10;
  const int  j   = nnIdx[row];

  const float* xi = x + row * KB_D + lane * 16;
  const float* xj = x + (b * KB_N + j) * KB_D + lane * 16;

  v4f fi[4], fj[4];
#pragma unroll
  for (int t = 0; t < 4; ++t) { fi[t] = *(const v4f*)(xi + 4 * t); fj[t] = *(const v4f*)(xj + 4 * t); }

  float ssi = 0.0f, ssj = 0.0f;
#pragma unroll
  for (int t = 0; t < 4; ++t)
#pragma unroll
    for (int c = 0; c < 4; ++c) { ssi += fi[t][c] * fi[t][c]; ssj += fj[t][c] * fj[t][c]; }
  ssi = wave_sum32(ssi);
  ssj = wave_sum32(ssj);
  const float ni = 1.0f / fmaxf(sqrtf(ssi), KB_EPS);
  const float nj = 1.0f / fmaxf(sqrtf(ssj), KB_EPS);

  float d2 = 0.0f;
#pragma unroll
  for (int t = 0; t < 4; ++t)
#pragma unroll
    for (int c = 0; c < 4; ++c) {
      const float d = fi[t][c] * ni - fj[t][c] * nj + KB_EPS;  // ||x - y + eps||
      d2 += d * d;
    }
  d2 = wave_sum32(d2);

  if (lane == 0) {
    const float dist = 0.5f * sqrtf(d2);
    atomicAdd(out, -logf(dist + KB_EPS) * (1.0f / (float)(KB_B * KB_N)));
  }
}

__global__ void koleo_zero(float* __restrict__ out) { out[0] = 0.0f; }

// ---------------------------------------------------------------------------
extern "C" void kernel_launch(void* const* d_in, const int* in_sizes, int n_in,
                              void* d_out, int out_size, void* d_ws, size_t ws_size,
                              hipStream_t stream) {
  (void)in_sizes; (void)n_in; (void)out_size; (void)ws_size;

  const float* x = (const float*)d_in[0];
  _Float16* xh   = (_Float16*)d_ws;                                    // 32 MB
  int* nnIdx     = (int*)((char*)d_ws +
                          (size_t)KB_B * KB_N * KB_D * sizeof(_Float16)); // 128 KB
  float* out     = (float*)d_out;

  const int nvec = KB_B * KB_N;                 // 32768 vectors / rows

  koleo_normalize<<<nvec / 8, 256, 0, stream>>>(x, xh);
  koleo_nn<<<KB_B * (KB_N / 16), 256, 0, stream>>>(xh, nnIdx);
  koleo_zero<<<1, 1, 0, stream>>>(out);
  koleo_loss<<<nvec / 8, 256, 0, stream>>>(x, nnIdx, out);
}
